// QDenseModel_38843684225834
// MI455X (gfx1250) — matrix operand order repaired and checked by reading, not compiled, not run on hardware
//
#include <hip/hip_runtime.h>

typedef _Float16 half_t;
typedef __attribute__((ext_vector_type(16))) _Float16 v16h;
typedef __attribute__((ext_vector_type(2)))  _Float16 h2v;
typedef __attribute__((ext_vector_type(8)))  float    v8f;
typedef __attribute__((ext_vector_type(4)))  float    f4v;
typedef __attribute__((ext_vector_type(4)))  unsigned u4v;

#define NLEV 127.0f
#define EPSQ 1e-8f
#define WPB  4
#define BLOCKSZ (WPB*32)
#define NSTEPS 16

// workspace float offsets
#define WS_ABSBITS 0
#define WS_PSF     1
#define WS_INV     2
#define WS_SF0     16
#define WS_SF1     80
#define WS_SF2     112
#define WS_SF3     144
#define WS_FS      160
#define WS_STREAM_BYTES 1024   // half_t stream: 16 steps * 32 lanes * 16 halves

__global__ void k_init(unsigned* wsu) { wsu[WS_ABSBITS] = 0u; }

__global__ void k_absmax(const f4v* __restrict__ x4, int n4, unsigned* __restrict__ wsu) {
    __shared__ float red[256];
    float m = 0.0f;
    for (int i = blockIdx.x * blockDim.x + threadIdx.x; i < n4; i += gridDim.x * blockDim.x) {
        f4v v = x4[i];
        m = fmaxf(m, fmaxf(fmaxf(fabsf(v[0]), fabsf(v[1])), fmaxf(fabsf(v[2]), fabsf(v[3]))));
    }
    red[threadIdx.x] = m;
    __syncthreads();
    for (int off = 128; off > 0; off >>= 1) {
        if (threadIdx.x < off) red[threadIdx.x] = fmaxf(red[threadIdx.x], red[threadIdx.x + off]);
        __syncthreads();
    }
    if (threadIdx.x == 0) atomicMax(wsu + WS_ABSBITS, __float_as_uint(red[0]));
}

__device__ inline float qclamp(float v) {  // round then clip to [-127,127]
    return fminf(fmaxf(rintf(v), -NLEV), NLEV);
}

// One block, 512 threads. Phase1: scales. Phase2: build f16 A-operand weight stream.
// Scale plan (all folded at prep time, power-of-two parts exact in f16):
//   L0 columns/bias:  w_int*2^-8, b_int*2^-8           -> acc0 == h~0 directly
//   L1,L2 columns:    w_int*sf_prev[f]  (2^8 * 2^-8 cancels), bias b_int*2^-8
//   L3 columns:       w_int*sf_prev[f]*2^8, bias b_int -> acc3 == integer-domain result
//   output:           acc3 * (sf3[o]*p_sf)
__global__ void k_prep(const float* __restrict__ w0, const float* __restrict__ b0,
                       const float* __restrict__ w1, const float* __restrict__ b1,
                       const float* __restrict__ w2, const float* __restrict__ b2,
                       const float* __restrict__ w3, const float* __restrict__ b3,
                       float* __restrict__ wsf, unsigned* __restrict__ wsu,
                       half_t* __restrict__ stream) {
    const int t = threadIdx.x;
    if (t == 144) {
        float am = __uint_as_float(wsu[WS_ABSBITS]);
        float s  = fmaxf(am / NLEV, EPSQ);
        wsf[WS_PSF] = s;
        wsf[WS_INV] = 1.0f / s;
    }
    if (t < 144) {
        const float* W; int K, O, sfo, o;
        if (t < 64)       { W = w0; K = 16; O = 64; sfo = WS_SF0; o = t;       }
        else if (t < 96)  { W = w1; K = 64; O = 32; sfo = WS_SF1; o = t - 64;  }
        else if (t < 128) { W = w2; K = 32; O = 32; sfo = WS_SF2; o = t - 96;  }
        else              { W = w3; K = 32; O = 5;  sfo = WS_SF3; o = t - 128; }
        float am = 0.0f;
        if (o < O) for (int k = 0; k < K; ++k) am = fmaxf(am, fabsf(W[o * K + k]));
        wsf[sfo + o] = (o < O) ? fmaxf(am / NLEV, EPSQ) : 1.0f;
    }
    __syncthreads();
    const float psf = wsf[WS_PSF];
    if (t < 5) wsf[WS_FS + t] = wsf[WS_SF3 + t] * psf;   // final dequant b_sf3

    // phase2: step s = t>>5, lane = t&31
    const int s = t >> 5, lane = t & 31;
    int L, ot, ks;
    if (s < 4)       { L = 0; ot = s;            ks = 0;            }
    else if (s < 10) { L = 1; ot = (s - 4) / 3;  ks = (s - 4) % 3;  }
    else if (s < 14) { L = 2; ot = (s - 10) / 2; ks = (s - 10) % 2; }
    else             { L = 3; ot = 0;            ks = s - 14;       }
    const float *W, *Bv, *sfR, *sfP; int K, Oreal;
    if (L == 0)      { W = w0; Bv = b0; K = 16; Oreal = 64; sfR = wsf + WS_SF0; sfP = wsf + WS_SF0; }
    else if (L == 1) { W = w1; Bv = b1; K = 64; Oreal = 32; sfR = wsf + WS_SF1; sfP = wsf + WS_SF0; }
    else if (L == 2) { W = w2; Bv = b2; K = 32; Oreal = 32; sfR = wsf + WS_SF2; sfP = wsf + WS_SF1; }
    else             { W = w3; Bv = b3; K = 32; Oreal = 5;  sfR = wsf + WS_SF3; sfP = wsf + WS_SF2; }
    const int o = ot * 16 + (lane & 15);
    const float DSH = 1.0f / 256.0f;
    half_t vals[16];
#pragma unroll
    for (int m = 0; m < 16; ++m) {
        // A-operand f16 layout: half m, lane l -> K index (contiguous: 16*(l>>4) + m):
        const int f = ks * 32 + (m + 16 * (lane >> 4));
        float val = 0.0f;
        if (o < Oreal) {
            const float sf = sfR[o];
            if (f < K) {
                const float wi = qclamp(W[o * K + f] / sf);
                if (L == 0)      val = wi * DSH;            // exact pow2 scale
                else if (L == 3) val = wi * (sfP[f] * 256.0f);
                else             val = wi * sfP[f];
            } else if (f == K) {
                const float bi = qclamp(Bv[o] / (sf * psf));  // quantized bias column
                val = (L == 3) ? bi : bi * DSH;               // exact pow2 scale
            }
        }
        vals[m] = (half_t)val;
    }
    half_t* dst = stream + (size_t)(s * 32 + lane) * 16;
#pragma unroll
    for (int m = 0; m < 16; ++m) dst[m] = vals[m];
}

__device__ inline void storeTile(h2v* buf, v8f a, int pos) {
    union { h2v p[4]; u4v u; } pk;
#pragma unroll
    for (int i = 0; i < 4; ++i) {
        h2v tt; tt[0] = (half_t)a[2 * i]; tt[1] = (half_t)a[2 * i + 1];
        pk.p[i] = tt;
    }
    *(u4v*)&buf[pos] = pk.u;   // pos is a multiple of 4 -> aligned b128
}

__device__ inline v16h loadB(const h2v* buf, int pos) {
    union { u4v q[2]; v16h v; } u;
    u.q[0] = *(const u4v*)&buf[pos];
    u.q[1] = *(const u4v*)&buf[pos + 4];
    return u.v;
}

__global__ __launch_bounds__(BLOCKSZ)
void k_main(const float* __restrict__ x, float* __restrict__ out,
            const float* __restrict__ wsf, const half_t* __restrict__ stream,
            int Brows, int nTiles, int totalWaves) {
    __shared__ h2v shb[WPB][576 + 320 + 320];   // bufA(16x36) | bufB(16x20) | bufC(16x20)
    const int tid  = threadIdx.x;
    const int lane = tid & 31;
    const int wvi  = tid >> 5;
    const int gw   = blockIdx.x * WPB + wvi;
    const int r    = lane & 15;
    const int hh   = lane >> 4;

    const float inv_s = wsf[WS_INV];
    const float fs0 = wsf[WS_FS + 0], fs1 = wsf[WS_FS + 1], fs2 = wsf[WS_FS + 2],
                fs3 = wsf[WS_FS + 3], fs4 = wsf[WS_FS + 4];

    // Load the whole pre-swizzled weight stream into registers (16 x v16h = 128 VGPRs)
    v16h wv[NSTEPS];
#pragma unroll
    for (int s = 0; s < NSTEPS; ++s) {
        union { u4v q[2]; v16h v; } u;
        const u4v* p = (const u4v*)(stream + (size_t)(s * 32 + lane) * 16);
        u.q[0] = p[0]; u.q[1] = p[1];
        wv[s] = u.v;
    }

    v16h bbias;  // constant B operand carrying the "1" row at K=0 (lanes<16 hold K=0..15)
#pragma unroll
    for (int m = 0; m < 16; ++m) bbias[m] = (half_t)0.0f;
    if (lane < 16) bbias[0] = (half_t)1.0f;

    v8f zc;
#pragma unroll
    for (int i = 0; i < 8; ++i) zc[i] = 0.0f;

    h2v* bufA = &shb[wvi][0];
    h2v* bufB = &shb[wvi][576];
    h2v* bufC = &shb[wvi][896];

    for (int tile = gw; tile < nTiles; tile += totalWaves) {
        const int row = tile * 16 + r;
        const bool rok = (row < Brows);
        if (lane < 16 && (tile + totalWaves) < nTiles)
            __builtin_prefetch(x + (size_t)(row + totalWaves * 16) * 16, 0, 1);

        // ---- build layer-0 B operand: quantized activations (+ bias-1 in K=16) ----
        v16h bx;
#pragma unroll
        for (int m = 0; m < 16; ++m) bx[m] = (half_t)0.0f;
        if (lane >= 16) bx[0] = (half_t)1.0f;          // K=16 bias column
        if (lane < 16 && rok) {
            const f4v* xr = (const f4v*)(x + (size_t)row * 16);
            f4v f[4]; f[0] = xr[0]; f[1] = xr[1]; f[2] = xr[2]; f[3] = xr[3];
#pragma unroll
            for (int i = 0; i < 16; ++i)
                bx[i] = (half_t)qclamp(f[i >> 2][i & 3] * inv_s);
        }

        // ---- layer 0: 16 -> 64 ----
        v8f a0[4];
#pragma unroll
        for (int ot = 0; ot < 4; ++ot)
            a0[ot] = __builtin_amdgcn_wmma_f32_16x16x32_f16(false, wv[ot], false, bx,
                                                            (short)0, zc, false, false);
#pragma unroll
        for (int ot = 0; ot < 4; ++ot)
            storeTile(bufA, a0[ot], r * 36 + ot * 8 + hh * 4);
        asm volatile("s_wait_dscnt 0x0" ::: "memory");

        // ---- layer 1: 64 -> 32 (2 data K-steps + bias step) ----
        v16h b10 = loadB(bufA, r * 36 + hh * 8);
        v16h b11 = loadB(bufA, r * 36 + 16 + hh * 8);
        v8f a1[2];
#pragma unroll
        for (int ot = 0; ot < 2; ++ot) {
            v8f c = __builtin_amdgcn_wmma_f32_16x16x32_f16(false, wv[4 + ot * 3 + 0], false, b10,
                                                           (short)0, zc, false, false);
            c = __builtin_amdgcn_wmma_f32_16x16x32_f16(false, wv[4 + ot * 3 + 1], false, b11,
                                                       (short)0, c, false, false);
            c = __builtin_amdgcn_wmma_f32_16x16x32_f16(false, wv[4 + ot * 3 + 2], false, bbias,
                                                       (short)0, c, false, false);
            a1[ot] = c;
        }
#pragma unroll
        for (int ot = 0; ot < 2; ++ot)
            storeTile(bufB, a1[ot], r * 20 + ot * 8 + hh * 4);
        asm volatile("s_wait_dscnt 0x0" ::: "memory");

        // ---- layer 2: 32 -> 32 ----
        v16h b2 = loadB(bufB, r * 20 + hh * 8);
        v8f a2[2];
#pragma unroll
        for (int ot = 0; ot < 2; ++ot) {
            v8f c = __builtin_amdgcn_wmma_f32_16x16x32_f16(false, wv[10 + ot * 2 + 0], false, b2,
                                                           (short)0, zc, false, false);
            c = __builtin_amdgcn_wmma_f32_16x16x32_f16(false, wv[10 + ot * 2 + 1], false, bbias,
                                                       (short)0, c, false, false);
            a2[ot] = c;
        }
#pragma unroll
        for (int ot = 0; ot < 2; ++ot)
            storeTile(bufC, a2[ot], r * 20 + ot * 8 + hh * 4);
        asm volatile("s_wait_dscnt 0x0" ::: "memory");

        // ---- layer 3: 32 -> 5 (padded to 16) ----
        v16h b3 = loadB(bufC, r * 20 + hh * 8);
        v8f a3 = __builtin_amdgcn_wmma_f32_16x16x32_f16(false, wv[14], false, b3,
                                                        (short)0, zc, false, false);
        a3 = __builtin_amdgcn_wmma_f32_16x16x32_f16(false, wv[15], false, bbias,
                                                    (short)0, a3, false, false);

        if (lane < 16 && rok) {
            float* op = out + (size_t)row * 5;
            op[0] = a3[0] * fs0; op[1] = a3[1] * fs1; op[2] = a3[2] * fs2;
            op[3] = a3[3] * fs3; op[4] = a3[4] * fs4;
        }
    }
}

extern "C" void kernel_launch(void* const* d_in, const int* in_sizes, int n_in,
                              void* d_out, int out_size, void* d_ws, size_t ws_size,
                              hipStream_t stream) {
    const float* x  = (const float*)d_in[0];
    const float* w0 = (const float*)d_in[1]; const float* b0 = (const float*)d_in[2];
    const float* w1 = (const float*)d_in[3]; const float* b1 = (const float*)d_in[4];
    const float* w2 = (const float*)d_in[5]; const float* b2 = (const float*)d_in[6];
    const float* w3 = (const float*)d_in[7]; const float* b3 = (const float*)d_in[8];
    float* out = (float*)d_out;

    const int N     = in_sizes[0];      // B*16
    const int Brows = N / 16;
    const int nTiles = (Brows + 15) / 16;

    float*    wsf    = (float*)d_ws;
    unsigned* wsu    = (unsigned*)d_ws;
    half_t*   wstrea = (half_t*)((char*)d_ws + WS_STREAM_BYTES);

    k_init<<<1, 1, 0, stream>>>(wsu);
    k_absmax<<<2048, 256, 0, stream>>>((const f4v*)x, N >> 2, wsu);
    k_prep<<<1, 512, 0, stream>>>(w0, b0, w1, b1, w2, b2, w3, b3, wsf, wsu, wstrea);

    int blocks = (nTiles + WPB - 1) / WPB;
    if (blocks > 2048) blocks = 2048;
    const int totalWaves = blocks * WPB;
    k_main<<<blocks, BLOCKSZ, 0, stream>>>(x, out, wsf, wstrea, Brows, nTiles, totalWaves);
}
